// SelfMultiheadAttention_60129542168
// MI455X (gfx1250) — matrix operand order repaired and checked by reading, not compiled, not run on hardware
//
#include <hip/hip_runtime.h>

typedef __attribute__((ext_vector_type(16))) __bf16 v16bf;
typedef __attribute__((ext_vector_type(2)))  __bf16 v2bf;
typedef __attribute__((ext_vector_type(8)))  float  v8f;
typedef __attribute__((ext_vector_type(8)))  unsigned int v8u;

#define EMBED   512
#define HEADS   8
#define HDIM    64
#define NGRAPH  32
#define NPG     128
#define NNODE   (NGRAPH * NPG)
#define QKVDIM  (3 * EMBED)

union BFrag { v8u u; v16bf b; };

// Pack two f32 -> 2x bf16 in one uint (v_cvt_pk_bf16_f32 when available).
static __device__ __forceinline__ unsigned int pk2bf(float lo, float hi) {
#if __has_builtin(__builtin_amdgcn_cvt_pk_bf16_f32)
    v2bf p = __builtin_amdgcn_cvt_pk_bf16_f32(lo, hi);
    return __builtin_bit_cast(unsigned int, p);
#else
    unsigned int ul = __builtin_bit_cast(unsigned int, lo);
    unsigned int uh = __builtin_bit_cast(unsigned int, hi);
    ul += 0x7FFFu + ((ul >> 16) & 1u);
    uh += 0x7FFFu + ((uh >> 16) & 1u);
    return (ul >> 16) | (uh & 0xFFFF0000u);
#endif
}

// Single f32 -> bf16 (low half of the packed convert).
static __device__ __forceinline__ unsigned short f2bf(float f) {
    return (unsigned short)(pk2bf(f, f) & 0xFFFFu);
}

// ---------------------------------------------------------------------------
// Pre-pass A: query f32 -> bf16, row-major (A-matrix K-pairs contiguous).
// ---------------------------------------------------------------------------
__global__ __launch_bounds__(256) void cvt_rowmajor_kernel(
    const float* __restrict__ src, unsigned int* __restrict__ dst, int n2)
{
    int i = blockIdx.x * blockDim.x + threadIdx.x;
    if (i < n2) {
        float2 v = *(const float2*)(src + 2 * (size_t)i);
        dst[i] = pk2bf(v.x, v.y);
    }
}

// ---------------------------------------------------------------------------
// Pre-pass B: W_in [K=512][N=1536] f32 -> Wt [N=1536][K=512] bf16 (transposed)
// so B-matrix fragments (fixed column n, contiguous K pair) are b32 loads.
// ---------------------------------------------------------------------------
__global__ __launch_bounds__(256) void cvt_w_transpose_kernel(
    const float* __restrict__ W, unsigned short* __restrict__ Wt)
{
    int i = blockIdx.x * blockDim.x + threadIdx.x;   // i < 512*1536
    int k = i / QKVDIM;
    int n = i % QKVDIM;
    Wt[(size_t)n * EMBED + k] = f2bf(W[i]);
}

// ---------------------------------------------------------------------------
// Kernel 1: qkv = query @ W_in + b_in   (bf16 WMMA, f32 accum)
// One wave computes a 16x64 output strip (4 N-tiles share one A fragment).
// Epilogue splits into Q (pre-scaled by Dh^-0.5), K, V bf16; wave-uniform
// base-pointer selection (each 64-col quad lies fully inside one of Q/K/V).
// ---------------------------------------------------------------------------
__global__ __launch_bounds__(256) void qkv_gemm_kernel(
    const unsigned short* __restrict__ Ab,  // [NNODE][EMBED] bf16
    const unsigned short* __restrict__ Wt,  // [QKVDIM][EMBED] bf16 (transposed)
    const float* __restrict__ bias,         // [QKVDIM]
    unsigned short* __restrict__ Qb,        // [NNODE][EMBED] bf16 (scaled)
    unsigned short* __restrict__ Kb,
    unsigned short* __restrict__ Vb)
{
    const int lane = threadIdx.x & 31;
    const int wid  = (blockIdx.x * blockDim.x + threadIdx.x) >> 5;
    const int NQ   = QKVDIM / 64;          // 24 column quads
    const int tm   = wid / NQ;             // row tile (16 rows)
    const int tq   = wid % NQ;             // column quad (64 cols)
    const int m    = lane & 15;
    const int half = lane >> 4;

    const unsigned short* arow = Ab + (size_t)(tm * 16 + m) * EMBED;

    v8f acc[4] = {};
    for (int kb = 0; kb < EMBED; kb += 32) {
        BFrag a;
#pragma unroll
        for (int v = 0; v < 8; ++v) {
            // A 16x32 bf16 layout: k = 16*(v/4) + 8*half + 2*(v%4)
            int ka = kb + ((v >> 2) << 4) + (half << 3) + ((v & 3) << 1);
            a.u[v] = *(const unsigned int*)(arow + ka);
        }
#pragma unroll
        for (int t = 0; t < 4; ++t) {
            const unsigned short* bcol = Wt + (size_t)(tq * 64 + t * 16 + m) * EMBED;
            BFrag b;
#pragma unroll
            for (int v = 0; v < 8; ++v) {
                // B 32x16 bf16 layout: k = 16*(v/4) + 8*half + 2*(v%4)
                int k = kb + ((v >> 2) << 4) + (half << 3) + ((v & 3) << 1);
                b.u[v] = *(const unsigned int*)(bcol + k);
            }
            acc[t] = __builtin_amdgcn_wmma_f32_16x16x32_bf16(false, a.b, false, b.b,
                                                             (short)0, acc[t], false, false);
        }
    }

    // Wave-uniform Q/K/V selection: which = (tq*64)/512
    const int which = tq >> 3;
    unsigned short* dst = (which == 0) ? Qb : (which == 1) ? Kb : Vb;
    const int colbase = tq * 64 - which * EMBED;
    const float scale = (which == 0) ? 0.125f : 1.0f;   // HDIM^-0.5 for Q

#pragma unroll
    for (int t = 0; t < 4; ++t) {
#pragma unroll
        for (int r = 0; r < 8; ++r) {
            int orow = tm * 16 + r + (half << 3);
            int ocol = tq * 64 + t * 16 + m;
            float val = (acc[t][r] + bias[ocol]) * scale;
            dst[(size_t)orow * EMBED + colbase + t * 16 + m] = f2bf(val);
        }
    }
}

// ---------------------------------------------------------------------------
// Kernel 2: fused attention, one workgroup per (graph, head), 8 waves.
// K tile and (transposed) V tile staged once into LDS per workgroup.
// Scores in 8 f32 WMMA accumulators; softmax via half-wave shfl_xor;
// P restaged through LDS (bf16) to re-enter the A-matrix layout for PV.
// Block-diagonal mask is structural (each block only sees its own graph).
// ---------------------------------------------------------------------------
__global__ __launch_bounds__(256) void attention_kernel(
    const unsigned short* __restrict__ Qb,
    const unsigned short* __restrict__ Kb,
    const unsigned short* __restrict__ Vb,
    float* __restrict__ out)                 // [NNODE][EMBED]
{
    __shared__ unsigned short Klds[NPG][HDIM + 2];     // [node][dim], padded
    __shared__ unsigned short Vt[HDIM][NPG + 2];       // [dim][node], padded
    __shared__ unsigned short Plds[8][16][NPG + 2];    // per-wave P, padded

    const int lane = threadIdx.x & 31;
    const int w    = threadIdx.x >> 5;     // wave 0..7
    const int g    = blockIdx.x >> 3;      // graph
    const int h    = blockIdx.x & 7;       // head
    const int m    = lane & 15;
    const int half = lane >> 4;
    const int hoff = h * HDIM;

    // ---- cooperative staging of K (row-major) and V (transposed) tiles
    for (int i = threadIdx.x; i < NPG * (HDIM / 2); i += 256) {
        int node = i >> 5;                 // / (HDIM/2)
        int dp   = (i & 31) << 1;          // dim pair
        *(unsigned int*)&Klds[node][dp] =
            *(const unsigned int*)(Kb + (size_t)(g * NPG + node) * EMBED + hoff + dp);
    }
    for (int i = threadIdx.x; i < NPG * HDIM; i += 256) {
        int node = i >> 6;                 // / HDIM
        int dim  = i & 63;
        Vt[dim][node] = Vb[(size_t)(g * NPG + node) * EMBED + hoff + dim];
    }
    __syncthreads();

    // ---- scores: S = Qs @ K^T   (16 x 128 per wave; Dh=64 => 2 k-steps)
    const unsigned short* qrow = Qb + (size_t)(g * NPG + w * 16 + m) * EMBED + hoff;
    v8f sacc[8] = {};
#pragma unroll
    for (int kk = 0; kk < 2; ++kk) {
        BFrag a;
#pragma unroll
        for (int v = 0; v < 8; ++v) {
            int k = kk * 32 + ((v >> 2) << 4) + (half << 3) + ((v & 3) << 1);
            a.u[v] = *(const unsigned int*)(qrow + k);
        }
#pragma unroll
        for (int t = 0; t < 8; ++t) {
            BFrag b;
#pragma unroll
            for (int v = 0; v < 8; ++v) {
                int k = kk * 32 + ((v >> 2) << 4) + (half << 3) + ((v & 3) << 1);
                b.u[v] = *(const unsigned int*)&Klds[t * 16 + m][k];
            }
            sacc[t] = __builtin_amdgcn_wmma_f32_16x16x32_bf16(false, a.b, false, b.b,
                                                              (short)0, sacc[t], false, false);
        }
    }

    // ---- row-wise softmax (row r+8*half lives in VGPR r across a 16-lane half)
    float rsum[8];
#pragma unroll
    for (int r = 0; r < 8; ++r) {
        float mx = sacc[0][r];
#pragma unroll
        for (int t = 1; t < 8; ++t) mx = fmaxf(mx, sacc[t][r]);
#pragma unroll
        for (int s = 1; s < 16; s <<= 1)
            mx = fmaxf(mx, __shfl_xor(mx, s, 32));
        float sum = 0.f;
#pragma unroll
        for (int t = 0; t < 8; ++t) {
            float e = __expf(sacc[t][r] - mx);
            sacc[t][r] = e;
            sum += e;
        }
#pragma unroll
        for (int s = 1; s < 16; s <<= 1)
            sum += __shfl_xor(sum, s, 32);
        rsum[r] = sum;
    }

    // ---- stage normalized P (bf16) into LDS row-major for the A layout
#pragma unroll
    for (int r = 0; r < 8; ++r) {
        float inv = 1.0f / rsum[r];
        int prow = r + (half << 3);
#pragma unroll
        for (int t = 0; t < 8; ++t) {
            Plds[w][prow][t * 16 + m] = f2bf(sacc[t][r] * inv);
        }
    }
    __syncthreads();

    // ---- O = P @ V   (16 x 64 per wave; K=128 => 4 k-steps of 32)
    v8f oacc[4] = {};
#pragma unroll
    for (int kk = 0; kk < 4; ++kk) {
        BFrag a;
#pragma unroll
        for (int v = 0; v < 8; ++v) {
            int k = kk * 32 + ((v >> 2) << 4) + (half << 3) + ((v & 3) << 1);
            a.u[v] = *(const unsigned int*)&Plds[w][m][k];
        }
#pragma unroll
        for (int t = 0; t < 4; ++t) {
            BFrag b;
#pragma unroll
            for (int v = 0; v < 8; ++v) {
                int k = kk * 32 + ((v >> 2) << 4) + (half << 3) + ((v & 3) << 1);   // node index
                b.u[v] = *(const unsigned int*)&Vt[t * 16 + m][k];
            }
            oacc[t] = __builtin_amdgcn_wmma_f32_16x16x32_bf16(false, a.b, false, b.b,
                                                              (short)0, oacc[t], false, false);
        }
    }

    // ---- write output [N, D], col = h*64 + d
#pragma unroll
    for (int t = 0; t < 4; ++t) {
#pragma unroll
        for (int r = 0; r < 8; ++r) {
            int orow = g * NPG + w * 16 + r + (half << 3);
            int ocol = hoff + t * 16 + m;
            out[(size_t)orow * EMBED + ocol] = oacc[t][r];
        }
    }
}

// ---------------------------------------------------------------------------
extern "C" void kernel_launch(void* const* d_in, const int* in_sizes, int n_in,
                              void* d_out, int out_size, void* d_ws, size_t ws_size,
                              hipStream_t stream) {
    const float* query = (const float*)d_in[0];   // [4096, 512]
    const float* W_in  = (const float*)d_in[1];   // [512, 1536]
    const float* b_in  = (const float*)d_in[2];   // [1536]
    // d_in[3] = batch_num_nodes: all graphs have exactly NPG nodes ->
    // block-diagonal mask realized structurally in attention_kernel.

    unsigned short* Qb = (unsigned short*)d_ws;               // 4 MB
    unsigned short* Kb = Qb + (size_t)NNODE * EMBED;          // 4 MB
    unsigned short* Vb = Kb + (size_t)NNODE * EMBED;          // 4 MB
    unsigned short* Ab = Vb + (size_t)NNODE * EMBED;          // 4 MB (query bf16)
    unsigned short* Wt = Ab + (size_t)NNODE * EMBED;          // 1.5 MB (W^T bf16)
    float* out = (float*)d_out;

    // Pre-pass conversions (memory-bound, one shot)
    {
        int n2 = NNODE * EMBED / 2;
        cvt_rowmajor_kernel<<<(n2 + 255) / 256, 256, 0, stream>>>(query, (unsigned int*)Ab, n2);
        int nw = EMBED * QKVDIM;
        cvt_w_transpose_kernel<<<(nw + 255) / 256, 256, 0, stream>>>(W_in, Wt);
    }

    // QKV projection: (4096/16) row tiles x (1536/64) col quads = 6144 waves
    {
        const int waves  = (NNODE / 16) * (QKVDIM / 64);
        qkv_gemm_kernel<<<waves / 8, 256, 0, stream>>>(Ab, Wt, b_in, Qb, Kb, Vb);
    }

    // Fused attention: one workgroup per (graph, head)
    attention_kernel<<<NGRAPH * HEADS, 256, 0, stream>>>(Qb, Kb, Vb, out);
}
